// Loss_19550691132030
// MI455X (gfx1250) — matrix operand order repaired and checked by reading, not compiled, not run on hardware
//
#include <hip/hip_runtime.h>
#include <hip/hip_bf16.h>
#include <math.h>

typedef __attribute__((ext_vector_type(16))) _Float16 v16h;
typedef __attribute__((ext_vector_type(8)))  _Float16 v8h;
typedef __attribute__((ext_vector_type(8)))  float    v8f;

namespace {
constexpr int NFR  = 64;      // B*T
constexpr int NJ   = 24;
constexpr int NV   = 6890;
constexpr int KP   = 207;     // pose-feature length
constexpr int KPP  = 224;     // padded to 7*32
constexpr int ROWS = NV * 3;  // 20670 rows of posedirs GEMM
constexpr int ROWT = 1292;    // ceil(20670/16) -> 20672 padded rows
constexpr int VT   = 431;     // ceil(6890/16) -> 6896 padded verts
constexpr int VPAD = 6896;
constexpr int MP   = 256;     // points per cloud (gen & merged)
constexpr int KF   = 232;     // FPS samples (MP - 24)
constexpr int VGRP = 54;      // ceil(VT/8)

// workspace byte offsets
constexpr size_t OFF_J0     = 0;                        // 72 f32
constexpr size_t OFF_GTR    = 512;                      // 64*24*9 f32
constexpr size_t OFF_PF     = OFF_GTR    + 55296;       // 2*64*224 f16
constexpr size_t OFF_PD16   = OFF_PF     + 57344;       // 20672*224 f16 (zero-padded)
constexpr size_t OFF_VPOSED = OFF_PD16   + 9261056;     // 2*20672*64 f32
constexpr size_t OFF_AMAT16 = OFF_VPOSED + 10584064;    // 2*64*32*16 f16
constexpr size_t OFF_LW16   = OFF_AMAT16 + 131072;      // 6896*32 f16
constexpr size_t OFF_VERTS  = OFF_LW16   + 441344;      // 2*64*6890*3 f32
constexpr size_t OFF_PSJ    = OFF_VERTS  + 10583040;    // 4608 f32
constexpr size_t OFF_MERGED = OFF_PSJ    + 18432;       // 64*256*3 f32
constexpr size_t OFF_CHAM   = OFF_MERGED + 196608;      // 64*2 f32
constexpr size_t OFF_PARTS  = OFF_CHAM   + 512;         // 3 f32
} // namespace

__device__ const int d_par[24] = {-1,0,0,0,1,2,3,4,5,6,7,8,9,9,9,12,13,14,16,17,18,19,20,21};

__device__ __forceinline__ v16h pack16(v8h lo, v8h hi) {
  return __builtin_shufflevector(lo, hi, 0, 1, 2, 3, 4, 5, 6, 7, 8, 9, 10, 11, 12, 13, 14, 15);
}

// ---------- lbs_weights -> f16, (6896 x 32) zero-padded ----------
__global__ void k_convert_lw(const float* __restrict__ lw, _Float16* __restrict__ lw16) {
  int i = blockIdx.x * blockDim.x + threadIdx.x;
  if (i >= VPAD * 32) return;
  int v = i >> 5, k = i & 31;
  float val = (v < NV && k < NJ) ? lw[v * NJ + k] : 0.0f;
  lw16[i] = (_Float16)val;
}

// ---------- posedirs -> f16, (20672 x 224) zero-padded ----------
__global__ void k_convert_pd(const float* __restrict__ pd, _Float16* __restrict__ pd16) {
  int i = blockIdx.x * blockDim.x + threadIdx.x;
  if (i >= ROWT * 16 * KPP) return;
  int r = i / KPP, k = i % KPP;
  float val = (r < ROWS && k < KP) ? pd[(size_t)r * KP + k] : 0.0f;
  pd16[i] = (_Float16)val;
}

// ---------- Rodrigues: pose (64,72) -> gt rotmats (64,24,9) ----------
__global__ void k_rodrigues(const float* __restrict__ pose, float* __restrict__ gtR) {
  int i = blockIdx.x * blockDim.x + threadIdx.x;
  if (i >= NFR * NJ) return;
  float ax = pose[i * 3 + 0], ay = pose[i * 3 + 1], az = pose[i * 3 + 2];
  float sq = ax * ax + ay * ay + az * az;
  float ang = sqrtf(sq + 1e-8f);
  float inv = 1.0f / ang;
  float x = ax * inv, y = ay * inv, z = az * inv;
  float s = sinf(ang), c = cosf(ang), ic = 1.0f - c;
  float* R = gtR + (size_t)i * 9;
  R[0] = 1.0f + ic * (-(y * y + z * z));
  R[1] = -s * z + ic * (x * y);
  R[2] =  s * y + ic * (x * z);
  R[3] =  s * z + ic * (x * y);
  R[4] = 1.0f + ic * (-(x * x + z * z));
  R[5] = -s * x + ic * (y * z);
  R[6] = -s * y + ic * (x * z);
  R[7] =  s * x + ic * (y * z);
  R[8] = 1.0f + ic * (-(x * x + y * y));
}

// ---------- pose features (rot[1:]-I) f16, zero-padded to 224 ----------
__global__ void k_posefeat(const float* __restrict__ predR, const float* __restrict__ gtR,
                           _Float16* __restrict__ pf) {
  int i = blockIdx.x * blockDim.x + threadIdx.x;
  if (i >= 2 * NFR * KPP) return;
  int s = i / (NFR * KPP);
  int r = i % (NFR * KPP);
  int n = r / KPP, k = r % KPP;
  float val = 0.0f;
  if (k < KP) {
    int j = k / 9 + 1, e = k % 9;
    const float* R = (s == 0) ? predR : gtR;
    val = R[((size_t)n * NJ + j) * 9 + e] - ((e == 0 || e == 4 || e == 8) ? 1.0f : 0.0f);
  }
  pf[i] = (_Float16)val;
}

// ---------- J0 = J_regressor @ v_template (frame independent, betas==0) ----------
__global__ void k_j0(const float* __restrict__ Jreg, const float* __restrict__ vt,
                     float* __restrict__ J0) {
  __shared__ float red[256];
  int o = blockIdx.x;          // j*3 + d, 72 outputs
  int j = o / 3, d = o % 3;
  float s = 0.0f;
  for (int v = threadIdx.x; v < NV; v += 256) s += Jreg[(size_t)j * NV + v] * vt[v * 3 + d];
  red[threadIdx.x] = s; __syncthreads();
  for (int st = 128; st > 0; st >>= 1) {
    if (threadIdx.x < st) red[threadIdx.x] += red[threadIdx.x + st];
    __syncthreads();
  }
  if (threadIdx.x == 0) J0[o] = red[0];
}

// ---------- kinematic chain -> per-joint 4x4 (flattened 16) in f16 ----------
__global__ void k_chain(const float* __restrict__ predR, const float* __restrict__ gtR,
                        const float* __restrict__ J0, _Float16* __restrict__ amat16) {
  int t = threadIdx.x;
  if (t >= 2 * NFR) return;
  int s = t >> 6, n = t & 63;
  const float* Rin = (s == 0) ? predR : gtR;
  float AR[NJ][9];
  float AT[NJ][3];
  for (int j = 0; j < NJ; ++j) {
    float Rj[9];
    for (int e = 0; e < 9; ++e) Rj[e] = Rin[((size_t)n * NJ + j) * 9 + e];
    int p = d_par[j];
    float rel[3];
    for (int d = 0; d < 3; ++d)
      rel[d] = J0[j * 3 + d] - ((p >= 0) ? J0[p * 3 + d] : 0.0f);
    if (j == 0) {
      for (int e = 0; e < 9; ++e) AR[0][e] = Rj[e];
      for (int d = 0; d < 3; ++d) AT[0][d] = rel[d];
    } else {
      for (int a = 0; a < 3; ++a) {
        for (int b = 0; b < 3; ++b) {
          float acc = 0.0f;
          for (int c = 0; c < 3; ++c) acc += AR[p][a * 3 + c] * Rj[c * 3 + b];
          AR[j][a * 3 + b] = acc;
        }
        float acc = AT[p][a];
        for (int c = 0; c < 3; ++c) acc += AR[p][a * 3 + c] * rel[c];
        AT[j][a] = acc;
      }
    }
  }
  _Float16* out = amat16 + ((size_t)s * NFR + n) * 512;   // 32 joints x 16 entries
  for (int j = 0; j < NJ; ++j) {
    for (int a = 0; a < 3; ++a) {
      float tc = 0.0f;
      for (int c = 0; c < 3; ++c) tc += AR[j][a * 3 + c] * J0[j * 3 + c];
      for (int b = 0; b < 3; ++b) out[j * 16 + a * 4 + b] = (_Float16)AR[j][a * 3 + b];
      out[j * 16 + a * 4 + 3] = (_Float16)(AT[j][a] - tc);
    }
    out[j * 16 + 12] = (_Float16)0.0f;
    out[j * 16 + 13] = (_Float16)0.0f;
    out[j * 16 + 14] = (_Float16)0.0f;
    out[j * 16 + 15] = (_Float16)1.0f;
  }
  for (int j = NJ; j < 32; ++j)
    for (int e = 0; e < 16; ++e) out[j * 16 + e] = (_Float16)0.0f;
}

// ---------- WMMA GEMM: v_posed = v_template + posedirs @ pose_feat^T ----------
// rows = (v*3+d) [pad 20672], cols = frame n [64]; one wave per 16x16 tile.
// All fragments are contiguous aligned v8h loads (b128) -- no scalarized guards.
__global__ void k_posed_gemm(const _Float16* __restrict__ pd16, const _Float16* __restrict__ pf,
                             const float* __restrict__ vt, float* __restrict__ vposed) {
  int lane = threadIdx.x & 31, w = threadIdx.x >> 5;
  int set = w >> 2, ct = w & 3;        // 8 waves: 2 sets x 4 col tiles
  int rt = blockIdx.x;                 // 0..1291
  int hs = lane >> 4, lm = lane & 15;
  int gr = rt * 16 + lm;               // A-matrix row (M = lane%16), padded buffer
  v8f c = {};
  const _Float16* arow = pd16 + (size_t)gr * KPP;
  const _Float16* brow = pf + ((size_t)set * NFR + (ct * 16 + lm)) * KPP;
#pragma unroll
  for (int kb = 0; kb < KPP; kb += 32) {
    __builtin_prefetch((const void*)(arow + kb + 64), 0, 1);
    v8h a0 = *(const v8h*)(arow + kb + hs * 8);
    v8h a1 = *(const v8h*)(arow + kb + 16 + hs * 8);
    v8h b0 = *(const v8h*)(brow + kb + hs * 8);
    v8h b1 = *(const v8h*)(brow + kb + 16 + hs * 8);
    v16h a = pack16(a0, a1);
    v16h b = pack16(b0, b1);
    c = __builtin_amdgcn_wmma_f32_16x16x32_f16(false, a, false, b, (short)0, c, false, false);
  }
  float* outp = vposed + (size_t)set * ROWT * 16 * NFR;
#pragma unroll
  for (int r = 0; r < 8; ++r) {
    int gr2 = rt * 16 + r + hs * 8;
    int n = ct * 16 + lm;
    if (gr2 < ROWS) outp[(size_t)gr2 * NFR + n] = vt[gr2] + c[r];
  }
}

// ---------- WMMA skinning: Tmat tile = lbs_weights(16x32) @ A(32x16); then verts ----------
// B staged transposed in LDS so the B fragment is two contiguous ds b128 loads.
__global__ void k_skin(const _Float16* __restrict__ lw16, const _Float16* __restrict__ amat16,
                       const float* __restrict__ vposed, float* __restrict__ verts) {
  __shared__ _Float16 sBt[512];   // [e][k] : e*32 + k
  __shared__ float sT[8 * 256];
  int blk = blockIdx.x;
  int set = blk / (NFR * VGRP);
  int rem = blk % (NFR * VGRP);
  int n = rem / VGRP, vg = rem % VGRP;
  const _Float16* am = amat16 + ((size_t)set * NFR + n) * 512;
  for (int i = threadIdx.x; i < 512; i += 256) {
    int k = i >> 4, e = i & 15;
    sBt[e * 32 + k] = am[i];
  }
  __syncthreads();
  int lane = threadIdx.x & 31, w = threadIdx.x >> 5;
  int hs = lane >> 4, lm = lane & 15;
  int vtile = vg * 8 + w;
  if (vtile < VT) {
    int v = vtile * 16 + lm;
    const _Float16* arow = lw16 + (size_t)v * 32;
    v8h a0 = *(const v8h*)(arow + hs * 8);
    v8h a1 = *(const v8h*)(arow + 16 + hs * 8);
    v8h b0 = *(const v8h*)(&sBt[lm * 32 + hs * 8]);
    v8h b1 = *(const v8h*)(&sBt[lm * 32 + 16 + hs * 8]);
    v16h a = pack16(a0, a1);
    v16h b = pack16(b0, b1);
    v8f c = {};
    c = __builtin_amdgcn_wmma_f32_16x16x32_f16(false, a, false, b, (short)0, c, false, false);
#pragma unroll
    for (int r = 0; r < 8; ++r) sT[w * 256 + (r + hs * 8) * 16 + lm] = c[r];
  }
  __syncthreads();
  if (vtile < VT && lane < 16) {
    int v = vtile * 16 + lm;
    if (v < NV) {
      const float* T = &sT[w * 256 + lm * 16];
      const float* vpb = vposed + (size_t)set * ROWT * 16 * NFR;
      float p0 = vpb[(size_t)(v * 3 + 0) * NFR + n];
      float p1 = vpb[(size_t)(v * 3 + 1) * NFR + n];
      float p2 = vpb[(size_t)(v * 3 + 2) * NFR + n];
      float* o = verts + (size_t)set * NFR * NV * 3 + ((size_t)n * NV + v) * 3;
#pragma unroll
      for (int a2 = 0; a2 < 3; ++a2)
        o[a2] = T[a2 * 4 + 0] * p0 + T[a2 * 4 + 1] * p1 + T[a2 * 4 + 2] * p2 + T[a2 * 4 + 3];
    }
  }
}

// ---------- predicted SMPL joints: J_regressor @ pred_verts ----------
__global__ void k_pjoint(const float* __restrict__ Jreg, const float* __restrict__ verts0,
                         float* __restrict__ psj) {
  __shared__ float red[256];
  int o = blockIdx.x;               // n*72 + j*3 + d  (== flat (n,24,3))
  int n = o / 72;
  int jd = o % 72;
  int j = jd / 3, d = jd % 3;
  float s = 0.0f;
  for (int v = threadIdx.x; v < NV; v += 256)
    s += Jreg[(size_t)j * NV + v] * verts0[((size_t)n * NV + v) * 3 + d];
  red[threadIdx.x] = s; __syncthreads();
  for (int st = 128; st > 0; st >>= 1) {
    if (threadIdx.x < st) red[threadIdx.x] += red[threadIdx.x + st];
    __syncthreads();
  }
  if (threadIdx.x == 0) psj[o] = red[0];
}

// ---------- FPS on GT verts + merge with GT joints (dists LDS-resident) ----------
__global__ void k_fps(const float* __restrict__ verts1, const float* __restrict__ gtJ,
                      float* __restrict__ merged) {
  __shared__ float sD[NV];
  __shared__ float rv[256];
  __shared__ int   ri[256];
  int n = blockIdx.x, tid = threadIdx.x;
  const float* pts = verts1 + (size_t)n * NV * 3;
  float* mout = merged + (size_t)n * MP * 3;
  for (int v = tid; v < NV; v += 256) sD[v] = 3.4e38f;
  __syncthreads();
  int last = 0;
  for (int k = 0; k < KF; ++k) {
    if (tid < 3) mout[k * 3 + tid] = pts[last * 3 + tid];
    float lx = pts[last * 3 + 0], ly = pts[last * 3 + 1], lz = pts[last * 3 + 2];
    float bv = -1.0f; int bi = 0;
    for (int v = tid; v < NV; v += 256) {
      float dx = pts[v * 3 + 0] - lx;
      float dy = pts[v * 3 + 1] - ly;
      float dz = pts[v * 3 + 2] - lz;
      float d = dx * dx + dy * dy + dz * dz;
      float nd = fminf(sD[v], d);
      sD[v] = nd;
      if (nd > bv) { bv = nd; bi = v; }
    }
    rv[tid] = bv; ri[tid] = bi; __syncthreads();
    for (int st = 128; st > 0; st >>= 1) {
      if (tid < st) {
        float ov = rv[tid + st]; int oi = ri[tid + st];
        if (ov > rv[tid] || (ov == rv[tid] && oi < ri[tid])) { rv[tid] = ov; ri[tid] = oi; }
      }
      __syncthreads();
    }
    last = ri[0];
    __syncthreads();
  }
  for (int j = tid; j < NJ; j += 256)
    for (int d = 0; d < 3; ++d)
      mout[(KF + j) * 3 + d] = gtJ[((size_t)n * NJ + j) * 3 + d];
}

// ---------- Chamfer-L1 per frame (both clouds LDS-resident) ----------
__global__ void k_chamfer(const float* __restrict__ gen, const float* __restrict__ merged,
                          float* __restrict__ cham) {
  __shared__ float xs[MP * 3];
  __shared__ float ys[MP * 3];
  __shared__ float red[256];
  int n = blockIdx.x, m = threadIdx.x;
  for (int d = 0; d < 3; ++d) {
    xs[m * 3 + d] = gen[((size_t)n * MP + m) * 3 + d];
    ys[m * 3 + d] = merged[((size_t)n * MP + m) * 3 + d];
  }
  __syncthreads();
  float x0 = xs[m * 3], x1 = xs[m * 3 + 1], x2 = xs[m * 3 + 2];
  float best = 3.4e38f;
  for (int g = 0; g < MP; ++g) {
    float d0 = x0 - ys[g * 3], d1 = x1 - ys[g * 3 + 1], d2 = x2 - ys[g * 3 + 2];
    best = fminf(best, d0 * d0 + d1 * d1 + d2 * d2);
  }
  float dxm = sqrtf(best + 1e-12f);
  float y0 = ys[m * 3], y1 = ys[m * 3 + 1], y2 = ys[m * 3 + 2];
  float best2 = 3.4e38f;
  for (int g = 0; g < MP; ++g) {
    float d0 = y0 - xs[g * 3], d1 = y1 - xs[g * 3 + 1], d2 = y2 - xs[g * 3 + 2];
    best2 = fminf(best2, d0 * d0 + d1 * d1 + d2 * d2);
  }
  float dym = sqrtf(best2 + 1e-12f);
  red[m] = dxm; __syncthreads();
  for (int st = 128; st > 0; st >>= 1) {
    if (m < st) red[m] += red[m + st];
    __syncthreads();
  }
  float s1 = red[0]; __syncthreads();
  red[m] = dym; __syncthreads();
  for (int st = 128; st > 0; st >>= 1) {
    if (m < st) red[m] += red[m + st];
    __syncthreads();
  }
  if (m == 0) { cham[n * 2] = s1; cham[n * 2 + 1] = red[0]; }
}

// ---------- generic single-block MSE ----------
__global__ void k_mse(const float* __restrict__ a, const float* __restrict__ b,
                      int count, float inv, float* __restrict__ out) {
  __shared__ float red[256];
  float s = 0.0f;
  for (int i = threadIdx.x; i < count; i += 256) {
    float d = a[i] - b[i];
    s += d * d;
  }
  red[threadIdx.x] = s; __syncthreads();
  for (int st = 128; st > 0; st >>= 1) {
    if (threadIdx.x < st) red[threadIdx.x] += red[threadIdx.x + st];
    __syncthreads();
  }
  if (threadIdx.x == 0) *out = red[0] * inv;
}

// ---------- finalize the 5 outputs ----------
__global__ void k_final(const float* __restrict__ parts, const float* __restrict__ cham,
                        float* __restrict__ out) {
  float s1 = 0.0f, s2 = 0.0f;
  for (int n = 0; n < NFR; ++n) { s1 += cham[n * 2]; s2 += cham[n * 2 + 1]; }
  float lc = 0.5f * (s1 / (float)(NFR * MP) + s2 / (float)(NFR * MP));
  float lp = parts[0], lsj = parts[1], lfj = parts[2];
  out[0] = lp + lsj + lfj + lc;
  out[1] = lp;
  out[2] = lsj;
  out[3] = lfj;
  out[4] = lc;
}

extern "C" void kernel_launch(void* const* d_in, const int* in_sizes, int n_in,
                              void* d_out, int out_size, void* d_ws, size_t ws_size,
                              hipStream_t stream) {
  const float* pose     = (const float*)d_in[1];
  const float* gtJ      = (const float*)d_in[2];   // full_joints (B,T,24,3)
  const float* predR    = (const float*)d_in[3];   // pred_rotmats (B,T,24,3,3)
  const float* predJ    = (const float*)d_in[4];   // pred_full_joints
  const float* gen      = (const float*)d_in[5];   // gen_points (B,T,256,3)
  const float* vt       = (const float*)d_in[6];   // v_template (V,3)
  const float* pd       = (const float*)d_in[8];   // posedirs (V,3,207)
  const float* Jreg     = (const float*)d_in[9];   // J_regressor (24,V)
  const float* lw       = (const float*)d_in[10];  // lbs_weights (V,24)
  float* out = (float*)d_out;

  char* ws = (char*)d_ws;
  float*    J0     = (float*)(ws + OFF_J0);
  float*    gtR    = (float*)(ws + OFF_GTR);
  _Float16* pf     = (_Float16*)(ws + OFF_PF);
  _Float16* pd16   = (_Float16*)(ws + OFF_PD16);
  float*    vposed = (float*)(ws + OFF_VPOSED);
  _Float16* amat16 = (_Float16*)(ws + OFF_AMAT16);
  _Float16* lw16   = (_Float16*)(ws + OFF_LW16);
  float*    verts  = (float*)(ws + OFF_VERTS);
  float*    psj    = (float*)(ws + OFF_PSJ);
  float*    merged = (float*)(ws + OFF_MERGED);
  float*    cham   = (float*)(ws + OFF_CHAM);
  float*    parts  = (float*)(ws + OFF_PARTS);
  const float* verts1 = verts + (size_t)NFR * NV * 3;  // GT set

  hipLaunchKernelGGL(k_convert_lw, dim3((VPAD * 32 + 255) / 256), dim3(256), 0, stream, lw, lw16);
  hipLaunchKernelGGL(k_convert_pd, dim3((ROWT * 16 * KPP + 255) / 256), dim3(256), 0, stream, pd, pd16);
  hipLaunchKernelGGL(k_rodrigues, dim3((NFR * NJ + 255) / 256), dim3(256), 0, stream, pose, gtR);
  hipLaunchKernelGGL(k_posefeat, dim3((2 * NFR * KPP + 255) / 256), dim3(256), 0, stream,
                     predR, gtR, pf);
  hipLaunchKernelGGL(k_j0, dim3(72), dim3(256), 0, stream, Jreg, vt, J0);
  hipLaunchKernelGGL(k_chain, dim3(1), dim3(128), 0, stream, predR, gtR, J0, amat16);
  hipLaunchKernelGGL(k_posed_gemm, dim3(ROWT), dim3(256), 0, stream, pd16, pf, vt, vposed);
  hipLaunchKernelGGL(k_skin, dim3(2 * NFR * VGRP), dim3(256), 0, stream, lw16, amat16, vposed, verts);
  hipLaunchKernelGGL(k_pjoint, dim3(NFR * 72), dim3(256), 0, stream, Jreg, verts, psj);
  hipLaunchKernelGGL(k_fps, dim3(NFR), dim3(256), 0, stream, verts1, gtJ, merged);
  hipLaunchKernelGGL(k_chamfer, dim3(NFR), dim3(256), 0, stream, gen, merged, cham);
  hipLaunchKernelGGL(k_mse, dim3(1), dim3(256), 0, stream, predR, gtR, NFR * NJ * 9,
                     1.0f / (float)(NFR * NJ * 9), parts + 0);
  hipLaunchKernelGGL(k_mse, dim3(1), dim3(256), 0, stream, psj, gtJ, NFR * NJ * 3,
                     1.0f / (float)(NFR * NJ * 3), parts + 1);
  hipLaunchKernelGGL(k_mse, dim3(1), dim3(256), 0, stream, predJ, gtJ, NFR * NJ * 3,
                     1.0f / (float)(NFR * NJ * 3), parts + 2);
  hipLaunchKernelGGL(k_final, dim3(1), dim3(1), 0, stream, parts, cham, out);
}
